// ConservativeMPLayer_91070486545075
// MI455X (gfx1250) — matrix operand order repaired and checked by reading, not compiled, not run on hardware
//
#include <hip/hip_runtime.h>
#include <hip/hip_bf16.h>
#include <math.h>

#define N_NODES 50000
#define N_EDGES 500000

typedef _Float16 v16h __attribute__((ext_vector_type(16)));
typedef float    v8f  __attribute__((ext_vector_type(8)));

// ---- weight-fragment region starts (units of one 32x16 fp16 fragment = 512 halves = 1KB) ----
#define F_PN_W2 0     // phi_node W2 128x128 : 4x8  = 32 frags
#define F_P1_W1 32    // phi1 W1     128x64  : 4x4  = 16
#define F_P2_W1 48    // phi2 W1     128x64  : 16
#define F_P1_W2 64    // phi1 W2     64x128  : 2x8  = 16
#define F_P2_W2 80    // phi2 W2     64x128  : 16
#define F_PE_W2 96    // phi_edge W2 64x64   : 2x4  = 8
#define F_PM_W1 104   // phi_msg W1  192x256 : 6x16 = 96
#define F_PM_W2 200   // phi_msg W2  256x128 : 8x8  = 64
#define F_PR_W1 264   // psi_rho  W1 128x64  : 16
#define F_PU_W1 280   // psi_rhou W1 128x64  : 16
#define F_PS_W1 296   // psi_e    W1 128x64  : 16   (total 312 frags = 319,488 B)

static constexpr size_t G_OFF     = 0;                                   // g : N x 128 f32
static constexpr size_t DELTA_OFF = (size_t)N_NODES * 128 * 4;           // 25,600,000
static constexpr size_t FRAG_OFF  = DELTA_OFF + (size_t)N_NODES * 4 * 4; // 26,400,000 (32B aligned)

// ---------------- device helpers ----------------
__device__ __forceinline__ float gelu_f(float x) {
  const float k = 0.7978845608028654f;  // sqrt(2/pi)
  return 0.5f * x * (1.0f + tanhf(k * (x + 0.044715f * x * x * x)));
}

__device__ __forceinline__ v8f splat8(float v) {
  v8f c;
#pragma unroll
  for (int r = 0; r < 8; ++r) c[r] = v;
  return c;
}

__device__ __forceinline__ v8f wmma_step(v8f c, v16h a, v16h b) {
  // D = A(16x32 f16) * B(32x16 f16) + C(16x16 f32)
  return __builtin_amdgcn_wmma_f32_16x16x32_f16(false, a, false, b, (short)0, c,
                                                false, false);
}

// A-fragment from f32 LDS tile [16 rows][ld cols], K-step kt (32 K-values).
// ISA 16-bit A layout: lanes 0-15 row M=lane, elems 0-7 => K 0-7, elems 8-15 => K 16-23;
// lanes 16-31 same row, K+8.
__device__ __forceinline__ v16h load_a_frag(const float* lds, int ld, int kt, int lane) {
  const int row = lane & 15;
  const int kb = kt * 32 + ((lane >> 4) & 1) * 8;
  v16h a;
#pragma unroll
  for (int h = 0; h < 16; ++h) {
    int k = kb + ((h & 8) << 1) + (h & 7);
    a[h] = (_Float16)lds[row * ld + k];
  }
  return a;
}

// B fragment pre-packed in ws: frag = 32 lanes x 16 halves, lane-major.
__device__ __forceinline__ v16h load_b_frag(const _Float16* fragBase, int frag, int lane) {
  const v16h* p = (const v16h*)(fragBase + (size_t)frag * 512);
  return p[lane];
}

// C tile (v8f: lane 0-15 col N=lane, rows 0-7; lane 16-31 rows 8-15) -> f32 LDS
__device__ __forceinline__ void store_c_lds(float* lds, int ld, int ntile, int lane,
                                            v8f c, bool act) {
  const int n = lane & 15;
  const int mb = ((lane >> 4) & 1) * 8;
#pragma unroll
  for (int r = 0; r < 8; ++r) {
    float v = c[r];
    if (act) v = gelu_f(v);
    lds[(mb + r) * ld + ntile * 16 + n] = v;
  }
}

// ---------------- weight packer: f32 row-major W[K][N] -> 32x16 f16 fragments ----------------
__global__ void pack_frags_kernel(const float* __restrict__ W, int Ntot,
                                  _Float16* __restrict__ out) {
  const int frag = blockIdx.x;
  const int lane = threadIdx.x;
  const int ntiles = Ntot >> 4;
  const int kt = frag / ntiles;
  const int nt = frag - kt * ntiles;
  const int n = nt * 16 + (lane & 15);
  const int kb = kt * 32 + ((lane >> 4) & 1) * 8;
  _Float16* o = out + (size_t)frag * 512 + lane * 16;
#pragma unroll
  for (int h = 0; h < 16; ++h) o[h] = (_Float16)W[(size_t)(kb + ((h & 8) << 1) + (h & 7)) * Ntot + n];
}

// ---------------- per-node kernel: h = phi_node(u); g = phi1(h)+phi2(h) ----------------
__global__ void __launch_bounds__(32)
node_kernel(const float* __restrict__ node_u,
            const float* __restrict__ pnW1, const float* __restrict__ pnB1,
            const float* __restrict__ pnB2,
            const float* __restrict__ p1B1, const float* __restrict__ p1B2,
            const float* __restrict__ p2B1, const float* __restrict__ p2B2,
            const _Float16* __restrict__ fragBase, float* __restrict__ g) {
  __shared__ float s0[16 * 128];
  __shared__ float s1[16 * 128];
  __shared__ float s2[16 * 64];
  __shared__ float s3[16 * 64];
  const int lane = threadIdx.x;
  const int base = blockIdx.x * 16;

  // phi_node L1: 4 -> 128, scalar (K=4 too small for WMMA), gelu
#pragma unroll 4
  for (int j = 0; j < 64; ++j) {
    int i = lane * 64 + j;
    int nd = i >> 7, col = i & 127;
    const float* u = node_u + (size_t)(base + nd) * 4;
    float acc = pnB1[col];
#pragma unroll
    for (int k = 0; k < 4; ++k) acc += u[k] * pnW1[k * 128 + col];
    s0[i] = gelu_f(acc);
  }
  __syncthreads();

  // phi_node L2: 128 -> 128, linear -> s1
  {
    v16h a[4];
#pragma unroll
    for (int kt = 0; kt < 4; ++kt) a[kt] = load_a_frag(s0, 128, kt, lane);
#pragma unroll
    for (int nt = 0; nt < 8; ++nt) {
      v8f c = splat8(pnB2[nt * 16 + (lane & 15)]);
#pragma unroll
      for (int kt = 0; kt < 4; ++kt)
        c = wmma_step(c, a[kt], load_b_frag(fragBase, F_PN_W2 + kt * 8 + nt, lane));
      store_c_lds(s1, 128, nt, lane, c, false);
    }
  }
  __syncthreads();

  // phi1/phi2 L1: 128 -> 64, gelu -> s2/s3
  {
    v16h a[4];
#pragma unroll
    for (int kt = 0; kt < 4; ++kt) a[kt] = load_a_frag(s1, 128, kt, lane);
#pragma unroll
    for (int nt = 0; nt < 4; ++nt) {
      v8f c = splat8(p1B1[nt * 16 + (lane & 15)]);
#pragma unroll
      for (int kt = 0; kt < 4; ++kt)
        c = wmma_step(c, a[kt], load_b_frag(fragBase, F_P1_W1 + kt * 4 + nt, lane));
      store_c_lds(s2, 64, nt, lane, c, true);
    }
#pragma unroll
    for (int nt = 0; nt < 4; ++nt) {
      v8f c = splat8(p2B1[nt * 16 + (lane & 15)]);
#pragma unroll
      for (int kt = 0; kt < 4; ++kt)
        c = wmma_step(c, a[kt], load_b_frag(fragBase, F_P2_W1 + kt * 4 + nt, lane));
      store_c_lds(s3, 64, nt, lane, c, true);
    }
  }
  __syncthreads();

  // g = (s2 @ phi1.W2 + b) + (s3 @ phi2.W2 + b) : 64 -> 128 each, accumulate in one C
  {
    v16h a2[2], a3[2];
    a2[0] = load_a_frag(s2, 64, 0, lane);
    a2[1] = load_a_frag(s2, 64, 1, lane);
    a3[0] = load_a_frag(s3, 64, 0, lane);
    a3[1] = load_a_frag(s3, 64, 1, lane);
    const int n = lane & 15;
    const int mb = ((lane >> 4) & 1) * 8;
#pragma unroll
    for (int nt = 0; nt < 8; ++nt) {
      v8f c = splat8(p1B2[nt * 16 + n] + p2B2[nt * 16 + n]);
      c = wmma_step(c, a2[0], load_b_frag(fragBase, F_P1_W2 + nt, lane));
      c = wmma_step(c, a2[1], load_b_frag(fragBase, F_P1_W2 + 8 + nt, lane));
      c = wmma_step(c, a3[0], load_b_frag(fragBase, F_P2_W2 + nt, lane));
      c = wmma_step(c, a3[1], load_b_frag(fragBase, F_P2_W2 + 8 + nt, lane));
#pragma unroll
      for (int r = 0; r < 8; ++r)
        g[(size_t)(base + mb + r) * 128 + nt * 16 + n] = c[r];
    }
  }
}

// ---------------- per-edge kernel: eps, phi_msg, heads, physics, scatter ----------------
__global__ void __launch_bounds__(32)
edge_kernel(const int* __restrict__ eidx, const float* __restrict__ eattr,
            const float* __restrict__ area, const float* __restrict__ g,
            const float* __restrict__ feW1, const float* __restrict__ feB1,
            const float* __restrict__ feB2,
            const float* __restrict__ pmB1, const float* __restrict__ pmB2,
            const float* __restrict__ prB1, const float* __restrict__ prW2,
            const float* __restrict__ prB2,
            const float* __restrict__ puB1, const float* __restrict__ puW2,
            const float* __restrict__ puB2,
            const float* __restrict__ qeB1, const float* __restrict__ qeW2,
            const float* __restrict__ qeB2,
            const float* __restrict__ sparam,
            const _Float16* __restrict__ fragBase, float* __restrict__ delta) {
  __shared__ float sA[16 * 192];  // [v | eps] input to phi_msg
  __shared__ float sH[16 * 256];  // hidden staging (also 3x 16x64 head regions)
  __shared__ float sM[16 * 128];  // m_ij
  __shared__ int   sSrc[16], sDst[16];
  __shared__ float sOut[16 * 4];  // a_rho, a_e, a_u0, a_u1 per edge
  const int lane = threadIdx.x;
  const int base = blockIdx.x * 16;

  if (lane < 16) {
    const int e = base + lane;
    sSrc[lane] = eidx[e];
    sDst[lane] = eidx[N_EDGES + e];
  }
  __syncthreads();

  // v = g[src] + g[dst] -> sA cols 0..127
#pragma unroll 4
  for (int j = 0; j < 64; ++j) {
    int i = lane * 64 + j;
    int ed = i >> 7, col = i & 127;
    sA[ed * 192 + col] =
        g[(size_t)sSrc[ed] * 128 + col] + g[(size_t)sDst[ed] * 128 + col];
  }
  // phi_edge L1: 1 -> 64 scalar, gelu -> sH (LD=64 staging)
#pragma unroll 4
  for (int j = 0; j < 32; ++j) {
    int i = lane * 32 + j;
    int ed = i >> 6, c = i & 63;
    float rr = eattr[(size_t)(base + ed) * 3 + 2];
    sH[ed * 64 + c] = gelu_f(rr * feW1[c] + feB1[c]);
  }
  __syncthreads();

  // eps = hidden @ phi_edge.W2 + b2 : 64 -> 64 -> sA cols 128..191
  {
    v16h a[2];
    a[0] = load_a_frag(sH, 64, 0, lane);
    a[1] = load_a_frag(sH, 64, 1, lane);
    const int n = lane & 15;
    const int mb = ((lane >> 4) & 1) * 8;
#pragma unroll
    for (int nt = 0; nt < 4; ++nt) {
      v8f c = splat8(feB2[nt * 16 + n]);
      c = wmma_step(c, a[0], load_b_frag(fragBase, F_PE_W2 + nt, lane));
      c = wmma_step(c, a[1], load_b_frag(fragBase, F_PE_W2 + 4 + nt, lane));
#pragma unroll
      for (int r = 0; r < 8; ++r)
        sA[(mb + r) * 192 + 128 + nt * 16 + n] = c[r];
    }
  }
  __syncthreads();

  // phi_msg L1: 192 -> 256, gelu -> sH (LD=256)
  {
    v16h a[6];
#pragma unroll
    for (int kt = 0; kt < 6; ++kt) a[kt] = load_a_frag(sA, 192, kt, lane);
#pragma unroll
    for (int nt = 0; nt < 16; ++nt) {
      v8f c = splat8(pmB1[nt * 16 + (lane & 15)]);
#pragma unroll
      for (int kt = 0; kt < 6; ++kt)
        c = wmma_step(c, a[kt], load_b_frag(fragBase, F_PM_W1 + kt * 16 + nt, lane));
      store_c_lds(sH, 256, nt, lane, c, true);
    }
  }
  __syncthreads();

  // phi_msg L2: 256 -> 128, linear -> sM
  {
    v16h a[8];
#pragma unroll
    for (int kt = 0; kt < 8; ++kt) a[kt] = load_a_frag(sH, 256, kt, lane);
#pragma unroll
    for (int nt = 0; nt < 8; ++nt) {
      v8f c = splat8(pmB2[nt * 16 + (lane & 15)]);
#pragma unroll
      for (int kt = 0; kt < 8; ++kt)
        c = wmma_step(c, a[kt], load_b_frag(fragBase, F_PM_W2 + kt * 8 + nt, lane));
      store_c_lds(sM, 128, nt, lane, c, false);
    }
  }
  __syncthreads();

  // psi heads L1: 128 -> 64 each, gelu -> sH regions {0:rho, 1:e, 2:rhou}
  {
    v16h a[4];
#pragma unroll
    for (int kt = 0; kt < 4; ++kt) a[kt] = load_a_frag(sM, 128, kt, lane);
    const int fr[3] = {F_PR_W1, F_PS_W1, F_PU_W1};
#pragma unroll
    for (int hd = 0; hd < 3; ++hd) {
      const float* b1 = (hd == 0) ? prB1 : ((hd == 1) ? qeB1 : puB1);
#pragma unroll
      for (int nt = 0; nt < 4; ++nt) {
        v8f c = splat8(b1[nt * 16 + (lane & 15)]);
#pragma unroll
        for (int kt = 0; kt < 4; ++kt)
          c = wmma_step(c, a[kt], load_b_frag(fragBase, fr[hd] + kt * 4 + nt, lane));
        store_c_lds(sH + hd * 1024, 64, nt, lane, c, true);
      }
    }
  }
  __syncthreads();

  // psi heads L2: 64 -> {1,1,2}, scalar dots (64 outputs over 32 lanes)
#pragma unroll
  for (int j = 0; j < 2; ++j) {
    int o = lane * 2 + j;
    int ed = o >> 2, w = o & 3;
    const float* hid;
    const float* w2;
    float bias;
    int stride, col;
    if (w == 0)      { hid = sH + ed * 64;        w2 = prW2; bias = prB2[0];     stride = 1; col = 0; }
    else if (w == 1) { hid = sH + 1024 + ed * 64; w2 = qeW2; bias = qeB2[0];     stride = 1; col = 0; }
    else             { hid = sH + 2048 + ed * 64; w2 = puW2; bias = puB2[w - 2]; stride = 2; col = w - 2; }
    float acc = bias;
#pragma unroll 8
    for (int k = 0; k < 64; ++k) acc += hid[k] * w2[k * stride + col];
    sOut[ed * 4 + w] = acc;
  }
  __syncthreads();

  // physics + conservative scatter (lanes 0..15, one edge each)
  if (lane < 16) {
    const int e = base + lane;
    const int src = sSrc[lane], dst = sDst[lane];
    const float dx = eattr[(size_t)e * 3 + 0];
    const float dy = eattr[(size_t)e * 3 + 1];
    const float rr = eattr[(size_t)e * 3 + 2];
    const float inv = 1.0f / (rr + 1e-12f);
    const float nx = dx * inv, ny = dy * inv;
    const float tx = -ny, ty = nx;
    const float a_rho = sOut[lane * 4 + 0];
    const float a_e   = sOut[lane * 4 + 1];
    const float au0   = sOut[lane * 4 + 2];
    const float au1   = sOut[lane * 4 + 3];
    const float nn = nx * nx + ny * ny;
    const float s_raw = a_rho * nn * rr;
    const float e_raw = a_e * nn * rr;
    const float vx = (au0 * nx + au1 * tx) * rr;
    const float vy = (au0 * ny + au1 * ty) * rr;
    const float dt = 0.015f / (1.0f + expf(-sparam[0]));
    const float maskf = (src < dst) ? 1.0f : 0.0f;
    const float isrc = maskf / area[src];
    const float idst = maskf / area[dst];
    atomicAdd(&delta[(size_t)src * 4 + 0], -dt * s_raw * isrc);
    atomicAdd(&delta[(size_t)src * 4 + 1], -dt * vx * isrc);
    atomicAdd(&delta[(size_t)src * 4 + 2], -dt * vy * isrc);
    atomicAdd(&delta[(size_t)src * 4 + 3], -dt * e_raw * isrc);
    atomicAdd(&delta[(size_t)dst * 4 + 0],  dt * s_raw * idst);
    atomicAdd(&delta[(size_t)dst * 4 + 1],  dt * vx * idst);
    atomicAdd(&delta[(size_t)dst * 4 + 2],  dt * vy * idst);
    atomicAdd(&delta[(size_t)dst * 4 + 3],  dt * e_raw * idst);
  }
}

// ---------------- small kernels ----------------
__global__ void zero_kernel(float* __restrict__ p, int n) {
  int i = blockIdx.x * 256 + threadIdx.x;
  if (i < n) p[i] = 0.0f;
}

__global__ void finalize_kernel(const float* __restrict__ node_u,
                                const float* __restrict__ delta,
                                const float* __restrict__ sparam,
                                float* __restrict__ out) {
  int i = blockIdx.x * 256 + threadIdx.x;
  if (i < N_NODES * 4) out[i] = node_u[i] + delta[i];
  if (i == 0) out[N_NODES * 4] = 0.015f / (1.0f + expf(-sparam[0]));
}

// ---------------- launch ----------------
extern "C" void kernel_launch(void* const* d_in, const int* in_sizes, int n_in,
                              void* d_out, int out_size, void* d_ws, size_t ws_size,
                              hipStream_t stream) {
  // inputs in setup_inputs() order; params flattened in sorted-key pytree order:
  // phi1, phi2, phi_edge, phi_msg, phi_node, psi_e, psi_rho, psi_rhou, s
  const float* node_u = (const float*)d_in[0];
  const int*   eidx   = (const int*)d_in[1];
  const float* eattr  = (const float*)d_in[2];
  const float* area   = (const float*)d_in[3];
  const float* p1W1 = (const float*)d_in[4];
  const float* p1B1 = (const float*)d_in[5];
  const float* p1W2 = (const float*)d_in[6];
  const float* p1B2 = (const float*)d_in[7];
  const float* p2W1 = (const float*)d_in[8];
  const float* p2B1 = (const float*)d_in[9];
  const float* p2W2 = (const float*)d_in[10];
  const float* p2B2 = (const float*)d_in[11];
  const float* feW1 = (const float*)d_in[12];
  const float* feB1 = (const float*)d_in[13];
  const float* feW2 = (const float*)d_in[14];
  const float* feB2 = (const float*)d_in[15];
  const float* pmW1 = (const float*)d_in[16];
  const float* pmB1 = (const float*)d_in[17];
  const float* pmW2 = (const float*)d_in[18];
  const float* pmB2 = (const float*)d_in[19];
  const float* pnW1 = (const float*)d_in[20];
  const float* pnB1 = (const float*)d_in[21];
  const float* pnW2 = (const float*)d_in[22];
  const float* pnB2 = (const float*)d_in[23];
  const float* qeW1 = (const float*)d_in[24];  // psi_e
  const float* qeB1 = (const float*)d_in[25];
  const float* qeW2 = (const float*)d_in[26];
  const float* qeB2 = (const float*)d_in[27];
  const float* prW1 = (const float*)d_in[28];  // psi_rho
  const float* prB1 = (const float*)d_in[29];
  const float* prW2 = (const float*)d_in[30];
  const float* prB2 = (const float*)d_in[31];
  const float* puW1 = (const float*)d_in[32];  // psi_rhou
  const float* puB1 = (const float*)d_in[33];
  const float* puW2 = (const float*)d_in[34];
  const float* puB2 = (const float*)d_in[35];
  const float* sparam = (const float*)d_in[36];

  float*    g     = (float*)((char*)d_ws + G_OFF);
  float*    delta = (float*)((char*)d_ws + DELTA_OFF);
  _Float16* frags = (_Float16*)((char*)d_ws + FRAG_OFF);

  // pack all GEMM weights into WMMA B-fragment layout (f32 -> f16), once per call
  pack_frags_kernel<<<32, 32, 0, stream>>>(pnW2, 128, frags + (size_t)F_PN_W2 * 512);
  pack_frags_kernel<<<16, 32, 0, stream>>>(p1W1,  64, frags + (size_t)F_P1_W1 * 512);
  pack_frags_kernel<<<16, 32, 0, stream>>>(p2W1,  64, frags + (size_t)F_P2_W1 * 512);
  pack_frags_kernel<<<16, 32, 0, stream>>>(p1W2, 128, frags + (size_t)F_P1_W2 * 512);
  pack_frags_kernel<<<16, 32, 0, stream>>>(p2W2, 128, frags + (size_t)F_P2_W2 * 512);
  pack_frags_kernel<<<8,  32, 0, stream>>>(feW2,  64, frags + (size_t)F_PE_W2 * 512);
  pack_frags_kernel<<<96, 32, 0, stream>>>(pmW1, 256, frags + (size_t)F_PM_W1 * 512);
  pack_frags_kernel<<<64, 32, 0, stream>>>(pmW2, 128, frags + (size_t)F_PM_W2 * 512);
  pack_frags_kernel<<<16, 32, 0, stream>>>(prW1,  64, frags + (size_t)F_PR_W1 * 512);
  pack_frags_kernel<<<16, 32, 0, stream>>>(puW1,  64, frags + (size_t)F_PU_W1 * 512);
  pack_frags_kernel<<<16, 32, 0, stream>>>(qeW1,  64, frags + (size_t)F_PS_W1 * 512);

  // per-node: h = phi_node(u); g = phi1(h) + phi2(h)   (N % 16 == 0)
  node_kernel<<<N_NODES / 16, 32, 0, stream>>>(node_u, pnW1, pnB1, pnB2,
                                               p1B1, p1B2, p2B1, p2B2, frags, g);

  zero_kernel<<<(N_NODES * 4 + 255) / 256, 256, 0, stream>>>(delta, N_NODES * 4);

  // per-edge fused pipeline  (E % 16 == 0)
  edge_kernel<<<N_EDGES / 16, 32, 0, stream>>>(eidx, eattr, area, g,
                                               feW1, feB1, feB2,
                                               pmB1, pmB2,
                                               prB1, prW2, prB2,
                                               puB1, puW2, puB2,
                                               qeB1, qeW2, qeB2,
                                               sparam, frags, delta);

  finalize_kernel<<<(N_NODES * 4 + 255) / 256, 256, 0, stream>>>(node_u, delta, sparam,
                                                                 (float*)d_out);
}
